// GATLayer_36051955482701
// MI455X (gfx1250) — matrix operand order repaired and checked by reading, not compiled, not run on hardware
//
#include <hip/hip_runtime.h>

#define BB 8
#define NN 1024
#define KDIM 256
#define ODIM 256
#define HH 4
#define DD 64
#define VROW 40   // padded LDS row stride (u16 units): 80B lane stride -> conflict-free b128 reads

#if defined(__has_builtin)
#if __has_builtin(__builtin_amdgcn_global_load_async_to_lds_b128) && \
    __has_builtin(__builtin_amdgcn_s_wait_asynccnt)
#define GAT_ASYNC 1
#endif
#endif

typedef __attribute__((ext_vector_type(16))) __bf16 v16bf;
typedef __attribute__((ext_vector_type(8)))  float  v8f;
typedef __attribute__((ext_vector_type(4)))  int    v4i;

union FragA { v16bf v; unsigned int w[8]; uint4 q[2]; };
union FragC { v8f  v; float f[8]; };

__device__ __forceinline__ unsigned int pkbf(float lo, float hi) {
    unsigned short l = __builtin_bit_cast(unsigned short, (__bf16)lo);
    unsigned short h = __builtin_bit_cast(unsigned short, (__bf16)hi);
    return (unsigned int)l | ((unsigned int)h << 16);
}
__device__ __forceinline__ float bf2f(unsigned short u) {
    union { unsigned int u; float f; } c; c.u = (unsigned int)u << 16; return c.f;
}

// ---------------------------------------------------------------------------
// Prep: x -> bf16 row-major ; W -> bf16 transposed (col-major, N x K).
// ---------------------------------------------------------------------------
__global__ __launch_bounds__(256) void gat_prep(
        const float* __restrict__ x, const float* __restrict__ W,
        unsigned short* __restrict__ xbf, unsigned short* __restrict__ wbT) {
    const int i = blockIdx.x * 256 + threadIdx.x;
    const long xi = (long)i * 4;
    if (xi < (long)BB * NN * KDIM) {
        float4 v = *(const float4*)(x + xi);
        *(uint2*)(xbf + xi) = make_uint2(pkbf(v.x, v.y), pkbf(v.z, v.w));
    }
    if (i < ODIM * KDIM / 4) {
        int c  = i >> 6;
        int kq = (i & 63) * 4;
        float w0 = W[(long)(kq + 0) * ODIM + c];
        float w1 = W[(long)(kq + 1) * ODIM + c];
        float w2 = W[(long)(kq + 2) * ODIM + c];
        float w3 = W[(long)(kq + 3) * ODIM + c];
        *(uint2*)(wbT + (long)c * KDIM + kq) = make_uint2(pkbf(w0, w1), pkbf(w2, w3));
    }
}

// ---------------------------------------------------------------------------
// Kernel 1: h = x @ W (M=8192, K=256, N=256), bf16 WMMA, f32 accumulate.
// h stored bf16 TRANSPOSED as (B, H, D, N). Every operand feed is b128.
// ---------------------------------------------------------------------------
__global__ __launch_bounds__(256) void gat_gemm_h(
        const unsigned short* __restrict__ xbf, const unsigned short* __restrict__ wbT,
        unsigned short* __restrict__ hbT) {
    const int lane = threadIdx.x & 31;
    const int wave = threadIdx.x >> 5;
    const int ct   = blockIdx.x;
    const int mt   = blockIdx.y * 8 + wave;
    const int half = lane >> 4;
    const int lr   = lane & 15;

    FragC acc;
#pragma unroll
    for (int v = 0; v < 8; ++v) acc.f[v] = 0.0f;

    const int colg = ct * 16 + lr;
    const unsigned short* xrow = xbf + (long)(mt * 16 + lr) * KDIM;
    const unsigned short* wcol = wbT + (long)colg * KDIM;

#pragma unroll
    for (int kc = 0; kc < KDIM / 32; ++kc) {
        FragA a, b;
        a.q[0] = *(const uint4*)(xrow + kc * 32 + 8 * half);
        a.q[1] = *(const uint4*)(xrow + kc * 32 + 16 + 8 * half);
        b.q[0] = *(const uint4*)(wcol + kc * 32 + 8 * half);
        b.q[1] = *(const uint4*)(wcol + kc * 32 + 16 + 8 * half);
        acc.v = __builtin_amdgcn_wmma_f32_16x16x32_bf16(
            false, a.v, false, b.v, (short)0, acc.v, false, false);
    }

    const int bh = (mt >> 6) * HH + (colg >> 6);
    const int d  = colg & 63;
    const int n0 = (mt & 63) * 16 + 8 * half;
    uint4 st;
    st.x = pkbf(acc.f[0], acc.f[1]);
    st.y = pkbf(acc.f[2], acc.f[3]);
    st.z = pkbf(acc.f[4], acc.f[5]);
    st.w = pkbf(acc.f[6], acc.f[7]);
    *(uint4*)(hbT + ((long)bh * DD + d) * NN + n0) = st;
}

// ---------------------------------------------------------------------------
// Kernel 2: attention logit halves e_i, e_j (coalesced over n).
// ---------------------------------------------------------------------------
__global__ __launch_bounds__(256) void gat_e(
        const unsigned short* __restrict__ hbT, const float* __restrict__ a,
        float* __restrict__ ei, float* __restrict__ ej) {
    int i = blockIdx.x * blockDim.x + threadIdx.x;
    if (i >= BB * HH * NN) return;
    int bh = i >> 10, n = i & (NN - 1), hh = bh & (HH - 1);
    const unsigned short* hp = hbT + (long)bh * DD * NN + n;
    const float* a1 = a + hh * 2 * DD;
    const float* a2 = a1 + DD;
    float s1 = 0.f, s2 = 0.f;
#pragma unroll 8
    for (int d = 0; d < DD; ++d) {
        float hv = bf2f(hp[(long)d * NN]);
        s1 += hv * a1[d]; s2 += hv * a2[d];
    }
    ei[i] = s1; ej[i] = s2;
}

// ---------------------------------------------------------------------------
// Kernel 3: fused masked-softmax attention (flash style), bf16 WMMA P.V.
// All 8 waves of a block share one (b,head): V chunks are staged once per
// block into LDS via async global->LDS (double buffered, ASYNCcnt), then
// consumed as conflict-free ds_load_b128 fragments.
// ---------------------------------------------------------------------------
__global__ __launch_bounds__(256) void gat_attn(
        const float* __restrict__ adj, const float* __restrict__ ei,
        const float* __restrict__ ej, const unsigned short* __restrict__ hbT,
        float* __restrict__ out) {
    __shared__ alignas(16) unsigned short vtile[2][DD * VROW];

    const int tid  = threadIdx.x;
    const int lane = tid & 31;
    const int wave = tid >> 5;
    const int bh   = blockIdx.x;            // 0..31 (shared by all waves in block)
    const int b    = bh >> 2, hh = bh & 3;
    const int nt   = blockIdx.y * 8 + wave; // 0..63
    const int half = lane >> 4;
    const int lr   = lane & 15;
    const int n    = nt * 16 + lr;

    const float  eiv = ei[bh * NN + n];
    const float* ejp = ej + bh * NN;
    const float* adjrow = adj + ((long)b * NN + n) * NN;

    // cooperative V copy: thread -> one b128 (d row = tid/4, k quad = tid%4)
    const int cpd = tid >> 2;
    const int cpk = (tid & 3) * 8;
    const unsigned short* gsrc = hbT + (long)bh * DD * NN + (long)cpd * NN + cpk;
    unsigned short* lsink0 = &vtile[0][cpd * VROW + cpk];
    unsigned short* lsink1 = &vtile[1][cpd * VROW + cpk];

#ifdef GAT_ASYNC
#define GAT_COPY(l, g)                                                              \
    __builtin_amdgcn_global_load_async_to_lds_b128(                                 \
        (__attribute__((address_space(1))) v4i*)(g),                                \
        (__attribute__((address_space(3))) v4i*)(l), 0, 0)
#define GAT_SETTLE() do { __builtin_amdgcn_s_wait_asynccnt(0); __syncthreads(); } while (0)
#else
#define GAT_COPY(l, g) (*(uint4*)(l) = *(const uint4*)(g))
#define GAT_SETTLE() __syncthreads()
#endif
#define GAT_STAGE(buf, mb)                                                          \
    do {                                                                            \
        unsigned short* _l = (buf) ? lsink1 : lsink0;                               \
        GAT_COPY(_l, gsrc + (mb));                                                  \
    } while (0)

    float m_run = -3.0e38f, l_run = 0.0f;
    FragC o0, o1, o2, o3;
#pragma unroll
    for (int v = 0; v < 8; ++v) { o0.f[v] = 0.f; o1.f[v] = 0.f; o2.f[v] = 0.f; o3.f[v] = 0.f; }

    GAT_STAGE(0, 0);
    GAT_SETTLE();

    for (int mt = 0; mt < NN / 32; ++mt) {
        const int mb  = mt * 32;
        const int buf = mt & 1;
        if (mt + 1 < NN / 32) GAT_STAGE(buf ^ 1, mb + 32);   // stream next chunk
        __builtin_prefetch(adjrow + mb + 32, 0, 0);

        // Scores in A-fragment order: per-lane K groups are 8-contiguous.
        float ejl[16], adl[16], sv[16];
        const float* ejb = ejp + mb + 8 * half;
        const float* ab  = adjrow + mb + 8 * half;
        *(float4*)(ejl +  0) = *(const float4*)(ejb +  0);
        *(float4*)(ejl +  4) = *(const float4*)(ejb +  4);
        *(float4*)(ejl +  8) = *(const float4*)(ejb + 16);
        *(float4*)(ejl + 12) = *(const float4*)(ejb + 20);
        *(float4*)(adl +  0) = *(const float4*)(ab  +  0);
        *(float4*)(adl +  4) = *(const float4*)(ab  +  4);
        *(float4*)(adl +  8) = *(const float4*)(ab  + 16);
        *(float4*)(adl + 12) = *(const float4*)(ab  + 20);
#pragma unroll
        for (int t = 0; t < 16; ++t) {
            float s = eiv + ejl[t];
            s = s > 0.f ? s : 0.2f * s;
            sv[t] = adl[t] > 0.f ? s : -9.0e15f;
        }

        // Online softmax: lanes r and r+16 mirror the same query row.
        float mloc = sv[0];
#pragma unroll
        for (int t = 1; t < 16; ++t) mloc = fmaxf(mloc, sv[t]);
        mloc = fmaxf(mloc, __shfl_xor(mloc, 16, 32));
        float m_new = fmaxf(m_run, mloc);
        float scale = __expf(m_run - m_new);

        float lsum = 0.f;
        FragA p;
#pragma unroll
        for (int j = 0; j < 8; ++j) {
            float p0 = __expf(sv[2 * j]     - m_new);
            float p1 = __expf(sv[2 * j + 1] - m_new);
            lsum += p0 + p1;
            p.w[j] = pkbf(p0, p1);
        }
        lsum += __shfl_xor(lsum, 16, 32);
        l_run = l_run * scale + lsum;
        m_run = m_new;

        // Rescale O: row r's factor lives in lane r; lane needs rows 8*half+v.
        float rs[8];
#pragma unroll
        for (int v = 0; v < 8; ++v) rs[v] = __shfl(scale, 8 * half + v, 32);
#pragma unroll
        for (int v = 0; v < 8; ++v) {
            o0.f[v] *= rs[v]; o1.f[v] *= rs[v]; o2.f[v] *= rs[v]; o3.f[v] *= rs[v];
        }

        // V fragments from LDS (padded stride -> conflict-free ds_load_b128).
        FragA v0f, v1f, v2f, v3f;
        const unsigned short* vt = &vtile[buf][8 * half];
        v0f.q[0] = *(const uint4*)(vt + ( 0 + lr) * VROW);
        v0f.q[1] = *(const uint4*)(vt + ( 0 + lr) * VROW + 16);
        v1f.q[0] = *(const uint4*)(vt + (16 + lr) * VROW);
        v1f.q[1] = *(const uint4*)(vt + (16 + lr) * VROW + 16);
        v2f.q[0] = *(const uint4*)(vt + (32 + lr) * VROW);
        v2f.q[1] = *(const uint4*)(vt + (32 + lr) * VROW + 16);
        v3f.q[0] = *(const uint4*)(vt + (48 + lr) * VROW);
        v3f.q[1] = *(const uint4*)(vt + (48 + lr) * VROW + 16);

        o0.v = __builtin_amdgcn_wmma_f32_16x16x32_bf16(false, p.v, false, v0f.v, (short)0, o0.v, false, false);
        o1.v = __builtin_amdgcn_wmma_f32_16x16x32_bf16(false, p.v, false, v1f.v, (short)0, o1.v, false, false);
        o2.v = __builtin_amdgcn_wmma_f32_16x16x32_bf16(false, p.v, false, v2f.v, (short)0, o2.v, false, false);
        o3.v = __builtin_amdgcn_wmma_f32_16x16x32_bf16(false, p.v, false, v3f.v, (short)0, o3.v, false, false);

        GAT_SETTLE();   // next buffer written & visible before anyone reads it
    }

    float linv = 1.0f / l_run;
    float rl[8];
#pragma unroll
    for (int v = 0; v < 8; ++v) rl[v] = __shfl(linv, 8 * half + v, 32);

#pragma unroll
    for (int v = 0; v < 8; ++v) {
        int  row  = v + 8 * half;
        long oidx = ((long)b * NN + nt * 16 + row) * ODIM + hh * DD;
        float s = rl[v];
        out[oidx +  0 + lr] = o0.f[v] * s;
        out[oidx + 16 + lr] = o1.f[v] * s;
        out[oidx + 32 + lr] = o2.f[v] * s;
        out[oidx + 48 + lr] = o3.f[v] * s;
    }
}

extern "C" void kernel_launch(void* const* d_in, const int* in_sizes, int n_in,
                              void* d_out, int out_size, void* d_ws, size_t ws_size,
                              hipStream_t stream) {
    const float* x   = (const float*)d_in[0];   // (B, N, 256)
    const float* adj = (const float*)d_in[1];   // (B, N, N)
    const float* W   = (const float*)d_in[2];   // (256, 256)
    const float* a   = (const float*)d_in[3];   // (H, 128)
    float* out = (float*)d_out;                 // (B, N, 256)

    unsigned short* xbf = (unsigned short*)d_ws;                  // 4 MB
    unsigned short* wbT = xbf + (long)BB * NN * KDIM;             // 128 KB
    unsigned short* hbT = wbT + (long)ODIM * KDIM;                // 4 MB
    float*          ei  = (float*)(hbT + (long)BB * NN * ODIM);   // 128 KB
    float*          ej  = ei + BB * HH * NN;                      // 128 KB

    gat_prep<<<(BB * NN * KDIM / 4 + 255) / 256, 256, 0, stream>>>(x, W, xbf, wbT);
    gat_gemm_h<<<dim3(ODIM / 16, 64), 256, 0, stream>>>(xbf, wbT, hbT);
    gat_e<<<(BB * HH * NN + 255) / 256, 256, 0, stream>>>(hbT, a, ei, ej);
    gat_attn<<<dim3(BB * HH, 8), 256, 0, stream>>>(adj, ei, ej, hbT, out);
}
